// AuxilNet_36447092474510
// MI455X (gfx1250) — matrix operand order repaired and checked by reading, not compiled, not run on hardware
//
#include <hip/hip_runtime.h>
#include <math.h>

// Problem constants (match the reference file)
#define NPILLARS 32768   // N = B * NPB
#define NPB      8192
#define MAXP     32
#define CF       64
// VOXEL_SIZE=(0.16,0.16,4.0), POINT_RANGE=(0.0,-39.68,-3.0)

typedef __bf16 bf16;
typedef __attribute__((ext_vector_type(8)))  __bf16 v8bf;
typedef __attribute__((ext_vector_type(16))) __bf16 v16bf;
typedef __attribute__((ext_vector_type(8)))  float  v8f;

// ---------------------------------------------------------------------------
// Kernel 1: per-pillar point-mean (u) and voxel center (k), stored as float4
// (xyz padded with 0) for clean 16B vector traffic later.
// NOTE: reference sums ALL 32 points (unmasked) and divides by num_points.
// ---------------------------------------------------------------------------
__global__ void prep_kernel(const float* __restrict__ voxels,
                            const int*   __restrict__ vnp,
                            const int*   __restrict__ coords,
                            float4* __restrict__ u_out,
                            float4* __restrict__ k_out)
{
    int i = blockIdx.x * blockDim.x + threadIdx.x;
    if (i >= NPILLARS) return;
    const float4* v = reinterpret_cast<const float4*>(voxels) + (size_t)i * MAXP;
    float sx = 0.f, sy = 0.f, sz = 0.f;
#pragma unroll
    for (int p = 0; p < MAXP; ++p) { float4 q = v[p]; sx += q.x; sy += q.y; sz += q.z; }
    float inv = 1.0f / (float)vnp[i];
    u_out[i] = make_float4(sx * inv, sy * inv, sz * inv, 0.f);

    int z = coords[i * 4 + 1], y = coords[i * 4 + 2], x = coords[i * 4 + 3];
    k_out[i] = make_float4((float)x * 0.16f + 0.08f,            // vx/2 + 0.0
                           (float)y * 0.16f + (0.08f - 39.68f), // vy/2 - 39.68
                           (float)z * 4.0f  + (2.0f  - 3.0f),   // vz/2 - 3.0
                           0.f);
}

// ---------------------------------------------------------------------------
// Kernel 2: Wcomb(4,64) = [W_cls; W_reg](4,64) @ W_fc(64,64), then emit the
// B matrix of the final GEMM *pre-swizzled into per-lane WMMA fragments*:
//   bfrag[ks][lane][j]  (2 x 32 x 16 bf16 = 2 KB)
// B(32x16 bf16, contiguous k, same order as A): lane n<16 holds column n, K = ks*32 + 0..15;
// lanes 16..31 hold column lane-16, K = ks*32 + 16..31. Columns 4..15 = 0.
// This lets every GEMM lane fetch its fragment with two b128 loads — no
// divergent predication in front of the WMMA (EXEC must stay all-ones).
// ---------------------------------------------------------------------------
__global__ void wcomb_kernel(const float* __restrict__ wfc,
                             const float* __restrict__ wcls,
                             const float* __restrict__ wreg,
                             bf16* __restrict__ bfrag)
{
    __shared__ float wc[4 * CF];
    int t = threadIdx.x;            // 256 = 4 rows * 64 cols
    int r = t >> 6, c = t & 63;
    float acc = 0.f;
    for (int j = 0; j < CF; ++j) {
        float wr = (r == 0) ? wcls[j] : wreg[(r - 1) * CF + j];
        acc += wr * wfc[j * CF + c];
    }
    wc[r * CF + c] = acc;           // Wcomb[r][c] = sum_j Wrow[r][j] * W_fc[j][c]
    __syncthreads();

    // 1024 fragment elements, 4 per thread; linear index vi == storage index.
#pragma unroll
    for (int q = 0; q < 4; ++q) {
        int vi   = t * 4 + q;
        int j    = vi & 15;
        int lane = (vi >> 4) & 31;
        int ks   = vi >> 9;
        int n    = lane & 15;
        int half = lane >> 4;
        int k    = ks * 32 + half * 16 + j;
        bfrag[vi] = (n < 4) ? (bf16)wc[n * CF + k] : (bf16)0.f;
    }
}

// ---------------------------------------------------------------------------
// Kernel 3: fused 3-NN search + inverse-distance interpolation.
// One thread per unknown point; known centers staged through LDS in 256-point
// tiles (no N^2 matrix materialized). Result p0 is written directly as bf16,
// row-major (N, 64), ready to be the WMMA A matrix.
// ---------------------------------------------------------------------------
__global__ void knn_interp_kernel(const float4* __restrict__ u,
                                  const float4* __restrict__ kc,
                                  const float*  __restrict__ feats,
                                  bf16* __restrict__ p0h)
{
    __shared__ float4 sk[256];
    const int b  = blockIdx.x >> 5;                       // NPB/256 = 32 blocks/batch
    const int li = ((blockIdx.x & 31) << 8) | threadIdx.x;
    const int gi = b * NPB + li;
    const float4 up = u[gi];

    float bd0 = 3e38f, bd1 = 3e38f, bd2 = 3e38f;
    int   bi0 = 0,     bi1 = 0,     bi2 = 0;

    for (int base = 0; base < NPB; base += 256) {
        __syncthreads();
        sk[threadIdx.x] = kc[b * NPB + base + threadIdx.x];
        __syncthreads();
#pragma unroll 4
        for (int t = 0; t < 256; ++t) {
            float4 kp = sk[t];
            float dx = up.x - kp.x, dy = up.y - kp.y, dz = up.z - kp.z;
            float d  = dx * dx + dy * dy + dz * dz;
            int idx  = base + t;
            if (d < bd2) {                         // strict '<' => first index wins ties
                if (d < bd1) {
                    bd2 = bd1; bi2 = bi1;
                    if (d < bd0) { bd1 = bd0; bi1 = bi0; bd0 = d; bi0 = idx; }
                    else         { bd1 = d;   bi1 = idx; }
                } else { bd2 = d; bi2 = idx; }
            }
        }
    }

    float r0 = 1.f / (sqrtf(fmaxf(bd0, 0.f)) + 1e-8f);
    float r1 = 1.f / (sqrtf(fmaxf(bd1, 0.f)) + 1e-8f);
    float r2 = 1.f / (sqrtf(fmaxf(bd2, 0.f)) + 1e-8f);
    float rs = 1.f / (r0 + r1 + r2);
    float w0 = r0 * rs, w1 = r1 * rs, w2 = r2 * rs;

    const float4* f0 = reinterpret_cast<const float4*>(feats + (size_t)(b * NPB + bi0) * CF);
    const float4* f1 = reinterpret_cast<const float4*>(feats + (size_t)(b * NPB + bi1) * CF);
    const float4* f2 = reinterpret_cast<const float4*>(feats + (size_t)(b * NPB + bi2) * CF);
    bf16* o = p0h + (size_t)gi * CF;
#pragma unroll
    for (int c = 0; c < CF / 4; ++c) {
        float4 a = f0[c], bb = f1[c], cc = f2[c];
        o[4 * c + 0] = (bf16)(w0 * a.x + w1 * bb.x + w2 * cc.x);
        o[4 * c + 1] = (bf16)(w0 * a.y + w1 * bb.y + w2 * cc.y);
        o[4 * c + 2] = (bf16)(w0 * a.z + w1 * bb.z + w2 * cc.z);
        o[4 * c + 3] = (bf16)(w0 * a.w + w1 * bb.w + w2 * cc.w);
    }
}

// ---------------------------------------------------------------------------
// Kernel 4: out(N,4) = p0(N,64) @ Wcomb(4,64).T via V_WMMA_F32_16X16X32_BF16.
// One wave per 16-row tile; K=64 -> two WMMA k-steps. All fragment loads are
// straight-line b128 vector loads (A from p0 rows, B from the pre-swizzled
// fragment table) — EXEC stays all-ones into the WMMAs.
// A (16x32 bf16): lane<16 row=lane, K in ks+0..15;
//                 lane>=16 row=lane-16, K in ks+16..31.
// D: VGPR v holds row tile_m + half*8 + v, column = lane&15.
// ---------------------------------------------------------------------------
__global__ void out_gemm_kernel(const bf16* __restrict__ p0h,
                                const bf16* __restrict__ bfrag,
                                float* __restrict__ out)
{
    const int wave   = blockIdx.x * (blockDim.x >> 5) + (threadIdx.x >> 5);
    const int lane   = threadIdx.x & 31;
    const int tile_m = wave * 16;
    if (tile_m >= NPILLARS) return;            // wave-uniform (never taken here)

    const int half = lane >> 4;                // 0: K-low half, 1: K-high half
    const int l    = lane & 15;                // A row within tile / B column

    // A row = 8 chunks of 8 bf16 (16 B each); pick chunks {2*half, 2*half+1} and
    // {4+2*half, 5+2*half} for the two k-steps.
    const v8bf* arow = reinterpret_cast<const v8bf*>(p0h + (size_t)(tile_m + l) * CF);
    v8bf a0l = arow[2 * half],     a0h = arow[2 * half + 1];
    v8bf a1l = arow[4 + 2 * half], a1h = arow[5 + 2 * half];

    // B fragments: bfrag viewed as v8bf chunks; ks=0 -> chunks [0,64), ks=1 -> [64,128)
    const v8bf* bfv = reinterpret_cast<const v8bf*>(bfrag);
    v8bf b0l = bfv[lane * 2],      b0h = bfv[lane * 2 + 1];
    v8bf b1l = bfv[64 + lane * 2], b1h = bfv[64 + lane * 2 + 1];

    v16bf a0 = __builtin_shufflevector(a0l, a0h, 0,1,2,3,4,5,6,7,8,9,10,11,12,13,14,15);
    v16bf a1 = __builtin_shufflevector(a1l, a1h, 0,1,2,3,4,5,6,7,8,9,10,11,12,13,14,15);
    v16bf b0 = __builtin_shufflevector(b0l, b0h, 0,1,2,3,4,5,6,7,8,9,10,11,12,13,14,15);
    v16bf b1 = __builtin_shufflevector(b1l, b1h, 0,1,2,3,4,5,6,7,8,9,10,11,12,13,14,15);

    v8f acc = {0.f, 0.f, 0.f, 0.f, 0.f, 0.f, 0.f, 0.f};
    acc = __builtin_amdgcn_wmma_f32_16x16x32_bf16(false, a0, false, b0,
                                                  (short)0, acc, false, false);
    acc = __builtin_amdgcn_wmma_f32_16x16x32_bf16(false, a1, false, b1,
                                                  (short)0, acc, false, false);

    // ---- D scatter: lane holds column l, rows tile_m + half*8 + v ----
    const int row0 = tile_m + half * 8;
    if (l == 0) {                                  // point_cls (N,1)
#pragma unroll
        for (int v = 0; v < 8; ++v) out[row0 + v] = acc[v];
    } else if (l < 4) {                            // point_reg (N,3)
#pragma unroll
        for (int v = 0; v < 8; ++v)
            out[(size_t)NPILLARS + (size_t)(row0 + v) * 3 + (l - 1)] = acc[v];
    }
}

// ---------------------------------------------------------------------------
// Launcher. Workspace layout (bytes):
//   [0, 512K)              u      : N float4
//   [512K, 1M)             k      : N float4
//   [1M, 1M+2K)            bfrag  : 2*32*16 bf16 (pre-swizzled WMMA B)
//   [1M+4K, 1M+4K+4M)      p0     : N*64 bf16    (total ~5.25 MB)
// ---------------------------------------------------------------------------
extern "C" void kernel_launch(void* const* d_in, const int* in_sizes, int n_in,
                              void* d_out, int out_size, void* d_ws, size_t ws_size,
                              hipStream_t stream)
{
    const float* voxels = (const float*)d_in[0];
    const int*   vnp    = (const int*)  d_in[1];
    const int*   coords = (const int*)  d_in[2];
    const float* feats  = (const float*)d_in[3];
    const float* wfc    = (const float*)d_in[4];
    const float* wcls   = (const float*)d_in[5];
    const float* wreg   = (const float*)d_in[6];
    float* out = (float*)d_out;

    char* ws = (char*)d_ws;
    float4* u     = (float4*)(ws);
    float4* kc    = (float4*)(ws + 16ull * NPILLARS);
    bf16*   bfrag = (bf16*)  (ws + 32ull * NPILLARS);
    bf16*   p0h   = (bf16*)  (ws + 32ull * NPILLARS + 4096);

    prep_kernel      <<<NPILLARS / 256, 256, 0, stream>>>(voxels, vnp, coords, u, kc);
    wcomb_kernel     <<<1,              256, 0, stream>>>(wfc, wcls, wreg, bfrag);
    knn_interp_kernel<<<NPILLARS / 256, 256, 0, stream>>>(u, kc, feats, p0h);
    out_gemm_kernel  <<<(NPILLARS / 16) / 8, 256, 0, stream>>>(p0h, bfrag, out);
}
